// Transformer_13889924235939
// MI455X (gfx1250) — matrix operand order repaired and checked by reading, not compiled, not run on hardware
//
#include <hip/hip_runtime.h>
#include <hip/hip_bf16.h>
#include <cstdint>

typedef __attribute__((ext_vector_type(16))) __bf16 v16bf;
typedef __attribute__((ext_vector_type(8)))  float  v8f;

#define DS_WAIT()    asm volatile("s_wait_dscnt 0" ::: "memory")
#define ASYNC_WAIT() asm volatile("s_wait_asynccnt 0" ::: "memory")

// 16B global -> LDS async copy (ASYNCcnt tracked); offset applies to both sides
#define ASYNC_CP16(ldsoff, gaddr, OFS)                                         \
    asm volatile("global_load_async_to_lds_b128 %0, %1, off offset:" #OFS      \
                 :: "v"(ldsoff), "v"(gaddr) : "memory")

static constexpr int Ec   = 768;
static constexpr int Hc   = 12;
static constexpr int DKc  = 64;
static constexpr int HIDc = 3072;
static constexpr int Lc   = 2;
static constexpr int Bc   = 2;
static constexpr int Sc   = 2048;
static constexpr int Mr   = Bc * Sc;    // 4096 activation rows
static constexpr int Vc   = 50257;

__device__ __forceinline__ __bf16 f2bf(float f) {
    unsigned u = __builtin_bit_cast(unsigned, f);
    u += 0x7FFFu + ((u >> 16) & 1u);
    unsigned short s = (unsigned short)(u >> 16);
    return __builtin_bit_cast(__bf16, s);
}

__device__ __forceinline__ v8f zero8() {
    v8f z;
#pragma unroll
    for (int i = 0; i < 8; ++i) z[i] = 0.0f;
    return z;
}

union FragAB { v16bf v; uint4 u[2]; };

// ---------------------------------------------------------------------------
// Weight transpose + f32 -> bf16 :  in [K][N] (batched) -> out [N][K]
// ---------------------------------------------------------------------------
__global__ void transpose_conv(const float* __restrict__ in, __bf16* __restrict__ out,
                               int K, int N) {
    const int z = blockIdx.z;
    in  += (size_t)z * K * N;
    out += (size_t)z * K * N;
    __shared__ float t[32][33];
    const int x = threadIdx.x & 31;
    const int y = threadIdx.x >> 5;          // 0..7
    const int bn = blockIdx.x * 32;
    const int bk = blockIdx.y * 32;
#pragma unroll
    for (int i = 0; i < 4; ++i) {
        int k = bk + y + i * 8;
        int n = bn + x;
        t[y + i * 8][x] = (k < K && n < N) ? in[(size_t)k * N + n] : 0.0f;
    }
    __syncthreads();
#pragma unroll
    for (int i = 0; i < 4; ++i) {
        int n = bn + y + i * 8;
        int k = bk + x;
        if (n < N && k < K) out[(size_t)n * K + k] = f2bf(t[x][y + i * 8]);
    }
}

// ---------------------------------------------------------------------------
// bf16 transpose for V: per batch b, in [S][768] -> out [768][S]
// ---------------------------------------------------------------------------
__global__ void transp_v(const __bf16* __restrict__ in, __bf16* __restrict__ out) {
    const int z = blockIdx.z;
    in  += (size_t)z * Sc * Ec;
    out += (size_t)z * Ec * Sc;
    __shared__ __bf16 t[32][33];
    const int x = threadIdx.x & 31;
    const int y = threadIdx.x >> 5;
    const int bc = blockIdx.x * 32;          // column (feature) tile
    const int bs = blockIdx.y * 32;          // seq tile
#pragma unroll
    for (int i = 0; i < 4; ++i)
        t[y + i * 8][x] = in[(size_t)(bs + y + i * 8) * Ec + bc + x];
    __syncthreads();
#pragma unroll
    for (int i = 0; i < 4; ++i)
        out[(size_t)(bc + y + i * 8) * Sc + bs + x] = t[x][y + i * 8];
}

// ---------------------------------------------------------------------------
// Token embedding + sinusoidal positions -> x (f32)
// ---------------------------------------------------------------------------
__global__ void embed_kernel(const int* __restrict__ tokens, const float* __restrict__ emb,
                             float* __restrict__ x) {
    const int row = blockIdx.x;              // 0..4095
    const int s   = row & (Sc - 1);
    const int tok = tokens[row];
#pragma unroll
    for (int j = 0; j < 3; ++j) {
        const int e = threadIdx.x + j * 256;
        const float angle = (float)s / __powf(10000.0f, 2.0f * (float)e / (float)Ec);
        const float pe = ((e & 1) == 0) ? __sinf(angle) : __cosf(angle);
        x[(size_t)row * Ec + e] = emb[(size_t)tok * Ec + e] + pe;
    }
}

// ---------------------------------------------------------------------------
// LayerNorm over E=768; writes f32 and bf16 copies
// ---------------------------------------------------------------------------
__global__ void layernorm_kernel(const float* __restrict__ x, const float* __restrict__ sc,
                                 const float* __restrict__ bi, float* __restrict__ hf,
                                 __bf16* __restrict__ hb) {
    const int row = blockIdx.x;
    const int tid = threadIdx.x;
    __shared__ float s1[256], s2[256];
    float a = 0.0f, b = 0.0f;
    float v[3];
#pragma unroll
    for (int j = 0; j < 3; ++j) {
        v[j] = x[(size_t)row * Ec + tid + j * 256];
        a += v[j];
        b += v[j] * v[j];
    }
    s1[tid] = a; s2[tid] = b;
    __syncthreads();
    for (int st = 128; st > 0; st >>= 1) {
        if (tid < st) { s1[tid] += s1[tid + st]; s2[tid] += s2[tid + st]; }
        __syncthreads();
    }
    const float mu  = s1[0] * (1.0f / Ec);
    const float var = s2[0] * (1.0f / Ec) - mu * mu;
    const float rs  = rsqrtf(var + 1e-5f);
#pragma unroll
    for (int j = 0; j < 3; ++j) {
        const int e = tid + j * 256;
        const float o = (v[j] - mu) * rs * sc[e] + bi[e];
        hf[(size_t)row * Ec + e] = o;
        hb[(size_t)row * Ec + e] = f2bf(o);
    }
}

// ---------------------------------------------------------------------------
// x = h + o  (residual add)
// ---------------------------------------------------------------------------
__global__ void add_kernel(const float* __restrict__ a, const float* __restrict__ b,
                           float* __restrict__ c) {
    const size_t i = (size_t)blockIdx.x * 256 + threadIdx.x;
    c[i] = a[i] + b[i];
}

// ---------------------------------------------------------------------------
// Tiled WMMA GEMM: C[M][N] = A[M][K](bf16) * Bt[N][K](bf16)^T (+bias)(+relu)(+resid)
// Block tile 128x128, 8 waves as 4x2, each wave 32x64 (2x4 WMMA tiles), K-step 32.
// LDS staging via GLOBAL_LOAD_ASYNC_TO_LDS_B128 (ASYNCcnt).
// N-edge: overshoot rows are loaded as garbage (in-workspace) and masked at store.
// ---------------------------------------------------------------------------
__global__ void __launch_bounds__(256)
gemm_bf16(const __bf16* __restrict__ A, const __bf16* __restrict__ Bt,
          const float* __restrict__ bias, const float* __restrict__ resid,
          float* __restrict__ Cf, __bf16* __restrict__ Cbf,
          int M, int N, int K, int relu) {
    __shared__ __bf16 As[128 * 32];          // 8 KB
    __shared__ __bf16 Bs[128 * 32];          // 8 KB

    const int tid  = threadIdx.x;
    const int lane = tid & 31;
    const int l16  = lane & 15;
    const int hi   = lane >> 4;
    const int wave = tid >> 5;               // 0..7
    const int wm   = wave & 3;               // 0..3 -> 32 rows each
    const int wn   = wave >> 2;              // 0..1 -> 64 cols each
    const int bm   = blockIdx.y * 128;
    const int bn   = blockIdx.x * 128;

    // staging assignment: each thread owns 32B of a 128-row x 64B tile
    const int sr = tid & 127;                // tile row
    const int sh = tid >> 7;                 // 32B half
    const unsigned lds_a = (unsigned)(uintptr_t)(As + sr * 32 + sh * 16);
    const unsigned lds_b = (unsigned)(uintptr_t)(Bs + sr * 32 + sh * 16);

    v8f acc[2][4];
#pragma unroll
    for (int i = 0; i < 2; ++i)
#pragma unroll
        for (int j = 0; j < 4; ++j) acc[i][j] = zero8();

    for (int k0 = 0; k0 < K; k0 += 32) {
        // ---- async stage A: 128 rows x 64B   (each thread 32B = 2 x b128)
        {
            const uint64_t ga =
                (uint64_t)(uintptr_t)(A + (size_t)(bm + sr) * K + k0 + sh * 16);
            ASYNC_CP16(lds_a, ga, 0);
            ASYNC_CP16(lds_a, ga, 16);
        }
        // ---- async stage B (rows are output columns)
        {
            const uint64_t gb =
                (uint64_t)(uintptr_t)(Bt + (size_t)(bn + sr) * K + k0 + sh * 16);
            ASYNC_CP16(lds_b, gb, 0);
            ASYNC_CP16(lds_b, gb, 16);
        }
        ASYNC_WAIT();
        __syncthreads();

        FragAB af[2], bf[4];
#pragma unroll
        for (int sm = 0; sm < 2; ++sm) {
            const uint4* p = (const uint4*)(As + (wm * 32 + sm * 16 + l16) * 32);
            af[sm].u[0] = p[hi];
            af[sm].u[1] = p[2 + hi];
        }
#pragma unroll
        for (int sn = 0; sn < 4; ++sn) {
            const uint4* p = (const uint4*)(Bs + (wn * 64 + sn * 16 + l16) * 32);
            bf[sn].u[0] = p[hi];
            bf[sn].u[1] = p[2 + hi];
        }
#pragma unroll
        for (int sm = 0; sm < 2; ++sm)
#pragma unroll
            for (int sn = 0; sn < 4; ++sn)
                acc[sm][sn] = __builtin_amdgcn_wmma_f32_16x16x32_bf16(
                    false, af[sm].v, false, bf[sn].v, (short)0, acc[sm][sn], false, false);
        __syncthreads();
    }

    // ---- epilogue
#pragma unroll
    for (int sm = 0; sm < 2; ++sm)
#pragma unroll
        for (int sn = 0; sn < 4; ++sn) {
            const int col = bn + wn * 64 + sn * 16 + l16;
            if (col >= N) continue;
            const float bia = bias ? bias[col] : 0.0f;
#pragma unroll
            for (int g = 0; g < 8; ++g) {
                const int row = bm + wm * 32 + sm * 16 + g + hi * 8;
                float v = acc[sm][sn][g] + bia;
                if (relu) v = fmaxf(v, 0.0f);
                if (resid) v += resid[(size_t)row * N + col];
                if (Cf)  Cf[(size_t)row * N + col]  = v;
                if (Cbf) Cbf[(size_t)row * N + col] = f2bf(v);
            }
        }
}

// ---------------------------------------------------------------------------
// Flash attention (causal). One wave = 16 query rows; block = 8 waves = 128 rows.
// grid = (S/128, B*H). q,k: [B*S][768] bf16 (head slice at h*64); vt: [B][768][S].
// Output o (f32) [B*S][768].
// ---------------------------------------------------------------------------
__global__ void attn_kernel(const __bf16* __restrict__ qb, const __bf16* __restrict__ kb,
                            const __bf16* __restrict__ vt, float* __restrict__ of32) {
    __shared__ float  Ssc[8][16 * 32];
    __shared__ float  rowm[8][16];
    __shared__ float  rowl[8][16];
    __shared__ float  rowsc[8][16];
    __shared__ __bf16 Pl[8][16 * 32];

    const int tid  = threadIdx.x;
    const int lane = tid & 31;
    const int l16  = lane & 15;
    const int hi   = lane >> 4;
    const int w    = tid >> 5;
    const int bh   = blockIdx.y;
    const int b    = bh / Hc;
    const int h    = bh % Hc;
    const int qbase = blockIdx.x * 128 + w * 16;

    // ---- load Q fragments (persist across kv loop): two K=32 halves of d=0..63
    FragAB qf[2];
    {
        const __bf16* qrow = qb + ((size_t)(b * Sc + qbase + l16)) * Ec + h * DKc;
#pragma unroll
        for (int kh = 0; kh < 2; ++kh) {
            const uint4* p0 = (const uint4*)(qrow + kh * 32 + hi * 8);
            const uint4* p1 = (const uint4*)(qrow + kh * 32 + 16 + hi * 8);
            qf[kh].u[0] = p0[0];
            qf[kh].u[1] = p1[0];
        }
    }

    v8f o[4];
#pragma unroll
    for (int i = 0; i < 4; ++i) o[i] = zero8();

    if (hi == 0) {
        rowm[w][l16] = -1e30f;
        rowl[w][l16] = 0.0f;
    }
    DS_WAIT();

    for (int j0 = 0; j0 < qbase + 16; j0 += 32) {
        // prefetch next K block
        if (j0 + 32 < qbase + 16)
            __builtin_prefetch(kb + ((size_t)(b * Sc + j0 + 32 + l16)) * Ec + h * DKc, 0, 0);

        // ---- S = Q * K^T for a 16x32 score tile (two 16x16 outputs, K=64)
        v8f sacc[2];
#pragma unroll
        for (int sn = 0; sn < 2; ++sn) {
            sacc[sn] = zero8();
            const int j = j0 + sn * 16 + l16;
            const __bf16* krow = kb + ((size_t)(b * Sc + j)) * Ec + h * DKc;
#pragma unroll
            for (int kh = 0; kh < 2; ++kh) {
                FragAB bfr;
                const uint4* p = (const uint4*)(krow + kh * 32 + hi * 8);
                bfr.u[0] = p[0];
                bfr.u[1] = p[2];
                sacc[sn] = __builtin_amdgcn_wmma_f32_16x16x32_bf16(
                    false, qf[kh].v, false, bfr.v, (short)0, sacc[sn], false, false);
            }
        }
        // ---- scale + causal mask, spill to wave-private LDS
#pragma unroll
        for (int sn = 0; sn < 2; ++sn)
#pragma unroll
            for (int g = 0; g < 8; ++g) {
                const int r = g + hi * 8;
                const int s = qbase + r;
                const int j = j0 + sn * 16 + l16;
                float v = sacc[sn][g] * 0.125f;     // 1/sqrt(64)
                if (j > s) v = -1e30f;
                Ssc[w][r * 32 + sn * 16 + l16] = v;
            }
        DS_WAIT();

        // ---- online softmax: lanes 0..15 each own one row
        if (hi == 0) {
            const int r = l16;
            float mold = rowm[w][r];
            float mnew = mold;
#pragma unroll
            for (int c = 0; c < 32; ++c) mnew = fmaxf(mnew, Ssc[w][r * 32 + c]);
            float lsum = 0.0f;
#pragma unroll
            for (int c = 0; c < 32; ++c) {
                const float p = __expf(Ssc[w][r * 32 + c] - mnew);
                Pl[w][r * 32 + c] = f2bf(p);
                lsum += p;
            }
            const float fac = __expf(mold - mnew);
            rowsc[w][r] = fac;
            rowl[w][r]  = rowl[w][r] * fac + lsum;
            rowm[w][r]  = mnew;
        }
        DS_WAIT();

        // ---- rescale accumulators
        float fs[8];
#pragma unroll
        for (int g = 0; g < 8; ++g) fs[g] = rowsc[w][g + hi * 8];
#pragma unroll
        for (int sn = 0; sn < 4; ++sn)
#pragma unroll
            for (int g = 0; g < 8; ++g) o[sn][g] *= fs[g];

        // ---- O += P * V   (P: 16x32 from LDS; V^T rows contiguous in seq)
        FragAB pf;
        {
            const uint4* p = (const uint4*)(&Pl[w][l16 * 32]);
            pf.u[0] = p[hi];
            pf.u[1] = p[2 + hi];
        }
#pragma unroll
        for (int sn = 0; sn < 4; ++sn) {
            const int d = sn * 16 + l16;
            FragAB vf;
            const uint4* p =
                (const uint4*)(vt + ((size_t)(b * Ec + h * DKc + d)) * Sc + j0 + hi * 8);
            vf.u[0] = p[0];
            vf.u[1] = p[2];
            o[sn] = __builtin_amdgcn_wmma_f32_16x16x32_bf16(
                false, pf.v, false, vf.v, (short)0, o[sn], false, false);
        }
    }

    // ---- normalize by row sums and store
    if (hi == 0) rowsc[w][l16] = 1.0f / rowl[w][l16];
    DS_WAIT();
    float fs[8];
#pragma unroll
    for (int g = 0; g < 8; ++g) fs[g] = rowsc[w][g + hi * 8];
#pragma unroll
    for (int sn = 0; sn < 4; ++sn)
#pragma unroll
        for (int g = 0; g < 8; ++g) {
            const int s   = qbase + g + hi * 8;
            const int col = h * DKc + sn * 16 + l16;
            of32[((size_t)(b * Sc + s)) * Ec + col] = o[sn][g] * fs[g];
        }
}

// ---------------------------------------------------------------------------
// Host-side orchestration
// ---------------------------------------------------------------------------
extern "C" void kernel_launch(void* const* d_in, const int* in_sizes, int n_in,
                              void* d_out, int out_size, void* d_ws, size_t ws_size,
                              hipStream_t stream) {
    (void)in_sizes; (void)n_in; (void)out_size; (void)ws_size;

    const int*   tokens = (const int*)d_in[0];
    const float* emb    = (const float*)d_in[1];
    const float* ln1_s  = (const float*)d_in[2];
    const float* ln1_b  = (const float*)d_in[3];
    const float* Wq     = (const float*)d_in[4];
    const float* bq     = (const float*)d_in[5];
    const float* Wk     = (const float*)d_in[6];
    const float* bk     = (const float*)d_in[7];
    const float* Wv     = (const float*)d_in[8];
    const float* bv     = (const float*)d_in[9];
    const float* ln2_s  = (const float*)d_in[10];
    const float* ln2_b  = (const float*)d_in[11];
    const float* W1     = (const float*)d_in[12];
    const float* b1     = (const float*)d_in[13];
    const float* W2     = (const float*)d_in[14];
    const float* b2     = (const float*)d_in[15];
    const float* Wout   = (const float*)d_in[16];
    const float* bout   = (const float*)d_in[17];
    float* out = (float*)d_out;

    // ---- workspace carve-up
    size_t off = 0;
    auto carve = [&](size_t bytes) -> void* {
        off = (off + 255) & ~(size_t)255;
        void* p = (char*)d_ws + off;
        off += bytes;
        return p;
    };
    __bf16* wqt   = (__bf16*)carve((size_t)Lc * Ec * Ec * 2);
    __bf16* wkt   = (__bf16*)carve((size_t)Lc * Ec * Ec * 2);
    __bf16* wvt   = (__bf16*)carve((size_t)Lc * Ec * Ec * 2);
    __bf16* w1t   = (__bf16*)carve((size_t)Lc * HIDc * Ec * 2);
    __bf16* w2t   = (__bf16*)carve((size_t)Lc * Ec * HIDc * 2);
    __bf16* woutt = (__bf16*)carve((size_t)Vc * Ec * 2);
    float*  xf32  = (float*)carve((size_t)Mr * Ec * 4);
    float*  hf32  = (float*)carve((size_t)Mr * Ec * 4);
    float*  of32  = (float*)carve((size_t)Mr * Ec * 4);
    __bf16* hbf   = (__bf16*)carve((size_t)Mr * Ec * 2);
    __bf16* qbf   = (__bf16*)carve((size_t)Mr * Ec * 2);
    __bf16* kbf   = (__bf16*)carve((size_t)Mr * Ec * 2);
    __bf16* vbf   = (__bf16*)carve((size_t)Mr * Ec * 2);
    __bf16* vtb   = (__bf16*)carve((size_t)Mr * Ec * 2);
    __bf16* abf   = (__bf16*)carve((size_t)Mr * HIDc * 2);
    __bf16* xbf   = (__bf16*)carve((size_t)Mr * Ec * 2);

    // ---- weight prep (transpose + bf16) : Bt layouts are [N][K]
    transpose_conv<<<dim3(2, 24, Lc * Hc), 256, 0, stream>>>(Wq, wqt, Ec, DKc);
    transpose_conv<<<dim3(2, 24, Lc * Hc), 256, 0, stream>>>(Wk, wkt, Ec, DKc);
    transpose_conv<<<dim3(2, 24, Lc * Hc), 256, 0, stream>>>(Wv, wvt, Ec, DKc);
    transpose_conv<<<dim3(96, 24, Lc), 256, 0, stream>>>(W1, w1t, Ec, HIDc);
    transpose_conv<<<dim3(24, 96, Lc), 256, 0, stream>>>(W2, w2t, HIDc, Ec);
    transpose_conv<<<dim3((Vc + 31) / 32, 24, 1), 256, 0, stream>>>(Wout, woutt, Ec, Vc);

    // ---- embedding + positions
    embed_kernel<<<Mr, 256, 0, stream>>>(tokens, emb, xf32);

    for (int l = 0; l < Lc; ++l) {
        layernorm_kernel<<<Mr, 256, 0, stream>>>(xf32, ln1_s + l * Ec, ln1_b + l * Ec,
                                                 hf32, hbf);
        // QKV projections (bias fused, bf16 outputs)
        gemm_bf16<<<dim3(6, 32), 256, 0, stream>>>(hbf, wqt + (size_t)l * Ec * Ec,
                                                   bq + l * Ec, nullptr, nullptr, qbf,
                                                   Mr, Ec, Ec, 0);
        gemm_bf16<<<dim3(6, 32), 256, 0, stream>>>(hbf, wkt + (size_t)l * Ec * Ec,
                                                   bk + l * Ec, nullptr, nullptr, kbf,
                                                   Mr, Ec, Ec, 0);
        gemm_bf16<<<dim3(6, 32), 256, 0, stream>>>(hbf, wvt + (size_t)l * Ec * Ec,
                                                   bv + l * Ec, nullptr, nullptr, vbf,
                                                   Mr, Ec, Ec, 0);
        // V -> V^T per batch for contiguous B-operand in P*V
        transp_v<<<dim3(24, 64, Bc), 256, 0, stream>>>(vbf, vtb);
        // causal flash attention
        attn_kernel<<<dim3(Sc / 128, Bc * Hc), 256, 0, stream>>>(qbf, kbf, vtb, of32);
        // x = h + o
        add_kernel<<<(Mr * Ec) / 256, 256, 0, stream>>>(hf32, of32, xf32);
        // MLP
        layernorm_kernel<<<Mr, 256, 0, stream>>>(xf32, ln2_s + l * Ec, ln2_b + l * Ec,
                                                 hf32, hbf);
        gemm_bf16<<<dim3(24, 32), 256, 0, stream>>>(hbf, w1t + (size_t)l * HIDc * Ec,
                                                    b1 + l * HIDc, nullptr, nullptr, abf,
                                                    Mr, HIDc, Ec, 1);
        // x = h2 + relu(h2 W1 + b1) W2 + b2   (residual fused; bf16 copy for final GEMM)
        gemm_bf16<<<dim3(6, 32), 256, 0, stream>>>(abf, w2t + (size_t)l * Ec * HIDc,
                                                   b2 + l * Ec, hf32, xf32, xbf,
                                                   Mr, Ec, HIDc, 0);
    }

    // ---- vocab projection (dominant GEMM): logits = x Wout + bout
    gemm_bf16<<<dim3((Vc + 127) / 128, 32), 256, 0, stream>>>(xbf, woutt, bout, nullptr,
                                                              out, nullptr, Mr, Vc, Ec, 0);
}
